// ScratchLinear_56581899157684
// MI455X (gfx1250) — compile-verified, matrix operand order checked by reading, not run on hardware
//
#include <hip/hip_runtime.h>
#include <cstdint>

// CDNA5 WMMA operand types (probe-confirmed signatures)
typedef __attribute__((ext_vector_type(16))) __bf16 v16bf;
typedef __attribute__((ext_vector_type(8)))  float  v8f;

#define DIM_K   2048
#define DIM_N   2048
#define M_TOTAL 8192          // 4 * 2048 rows, flattened
#define BM      128
#define BN      128
#define BK      32
#define KTILES  (DIM_K / BK)  // 64
// dwords per LDS row: 16 data dwords (32 bf16) + 4 pad = 20 -> rows stay 16B
// aligned (b128-friendly) and 20*l mod 64 is conflict-free for 16-row strides.
#define LDSW    20

union Frag { v16bf v; uint4 q[2]; };

// Split two fp32 into packed truncated-bf16 "hi" and packed bf16(f - hi) "lo".
// 6 VALU per 2 elements: 2x v_and, 2x v_sub_f32, 2x v_perm_b32.
__device__ __forceinline__ void split2(float f0, float f1, uint32_t& hi, uint32_t& lo) {
  uint32_t u0 = __builtin_bit_cast(uint32_t, f0);
  uint32_t u1 = __builtin_bit_cast(uint32_t, f1);
  // result[15:0] = u0[31:16], result[31:16] = u1[31:16]
  hi = __builtin_amdgcn_perm(u1, u0, 0x07060302u);
  float r0 = f0 - __builtin_bit_cast(float, u0 & 0xFFFF0000u);
  float r1 = f1 - __builtin_bit_cast(float, u1 & 0xFFFF0000u);
  lo = __builtin_amdgcn_perm(__builtin_bit_cast(uint32_t, r1),
                             __builtin_bit_cast(uint32_t, r0), 0x07060302u);
}

__global__ __launch_bounds__(256)
void ScratchLinear_wmma_bf16x3(const float* __restrict__ X,
                               const float* __restrict__ W,
                               const float* __restrict__ Bias,
                               float* __restrict__ Out) {
  // hi/lo bf16 planes for A (x tile) and B (W tile), double buffered. 80 KB.
  __shared__ __align__(16) uint32_t sAhi[2][BM * LDSW];
  __shared__ __align__(16) uint32_t sAlo[2][BM * LDSW];
  __shared__ __align__(16) uint32_t sBhi[2][BN * LDSW];
  __shared__ __align__(16) uint32_t sBlo[2][BN * LDSW];

  const int tid  = threadIdx.x;
  const int lane = tid & 31;        // wave32
  const int wave = tid >> 5;        // 0..7
  const int wm   = wave & 3;        // 4 waves along M
  const int wn   = wave >> 2;       // 2 waves along N
  const int n0   = blockIdx.x * BN;
  const int m0   = blockIdx.y * BM;

  // ---- global -> register staging: thread owns one half-row (16 floats)
  const int ldr = tid >> 1;         // 0..127 (row within tile)
  const int ldk = (tid & 1) * 16;   // k offset within BK
  const float* xg = X + (size_t)(m0 + ldr) * DIM_K + ldk;
  const float* wg = W + (size_t)(n0 + ldr) * DIM_K + ldk;

  float4 axr[4], bxr[4];

  auto gload = [&](int kt) {
    const float4* xp = (const float4*)(xg + kt * BK);
    const float4* wp = (const float4*)(wg + kt * BK);
#pragma unroll
    for (int c = 0; c < 4; ++c) { axr[c] = xp[c]; bxr[c] = wp[c]; }
  };

  auto lstore = [&](int buf) {
    uint32_t hA[8], lA[8], hB[8], lB[8];
#pragma unroll
    for (int c = 0; c < 4; ++c) {
      split2(axr[c].x, axr[c].y, hA[2 * c],     lA[2 * c]);
      split2(axr[c].z, axr[c].w, hA[2 * c + 1], lA[2 * c + 1]);
      split2(bxr[c].x, bxr[c].y, hB[2 * c],     lB[2 * c]);
      split2(bxr[c].z, bxr[c].w, hB[2 * c + 1], lB[2 * c + 1]);
    }
    const int base = ldr * LDSW + (ldk >> 1);   // dword index, multiple of 4
    *(uint4*)&sAhi[buf][base]     = make_uint4(hA[0], hA[1], hA[2], hA[3]);
    *(uint4*)&sAhi[buf][base + 4] = make_uint4(hA[4], hA[5], hA[6], hA[7]);
    *(uint4*)&sAlo[buf][base]     = make_uint4(lA[0], lA[1], lA[2], lA[3]);
    *(uint4*)&sAlo[buf][base + 4] = make_uint4(lA[4], lA[5], lA[6], lA[7]);
    *(uint4*)&sBhi[buf][base]     = make_uint4(hB[0], hB[1], hB[2], hB[3]);
    *(uint4*)&sBhi[buf][base + 4] = make_uint4(hB[4], hB[5], hB[6], hB[7]);
    *(uint4*)&sBlo[buf][base]     = make_uint4(lB[0], lB[1], lB[2], lB[3]);
    *(uint4*)&sBlo[buf][base + 4] = make_uint4(lB[4], lB[5], lB[6], lB[7]);
  };

  v8f acc[2][4];
#pragma unroll
  for (int mi = 0; mi < 2; ++mi)
#pragma unroll
    for (int nt = 0; nt < 4; ++nt) acc[mi][nt] = 0.0f;

  auto compute = [&](int buf) {
    // A fragments for this wave's two M tiles (hi and lo planes).
    // ISA A layout (16-bit, 16x32): lane&15 = row, lane>>4 selects K-half,
    // VGPR j=0..3 -> K = khalf*8 + 2j..2j+1, j=4..7 -> +16. Two b128 loads.
    Frag ahi[2], alo[2];
#pragma unroll
    for (int mi = 0; mi < 2; ++mi) {
      const int row  = wm * 32 + mi * 16 + (lane & 15);
      const int base = row * LDSW + ((lane >> 4) << 2);  // 16B aligned
      ahi[mi].q[0] = *(const uint4*)&sAhi[buf][base];
      ahi[mi].q[1] = *(const uint4*)&sAhi[buf][base + 8];
      alo[mi].q[0] = *(const uint4*)&sAlo[buf][base];
      alo[mi].q[1] = *(const uint4*)&sAlo[buf][base + 8];
    }
#pragma unroll
    for (int nt = 0; nt < 4; ++nt) {
      Frag bhi, blo;
      const int row  = wn * 64 + nt * 16 + (lane & 15);
      const int base = row * LDSW + ((lane >> 4) << 2);
      bhi.q[0] = *(const uint4*)&sBhi[buf][base];
      bhi.q[1] = *(const uint4*)&sBhi[buf][base + 8];
      blo.q[0] = *(const uint4*)&sBlo[buf][base];
      blo.q[1] = *(const uint4*)&sBlo[buf][base + 8];
#pragma unroll
      for (int mi = 0; mi < 2; ++mi) {
        // bf16x3 fp32 emulation: hi*hi + lo*hi + hi*lo, f32 accumulate
        acc[mi][nt] = __builtin_amdgcn_wmma_f32_16x16x32_bf16(
            false, ahi[mi].v, false, bhi.v, (short)0, acc[mi][nt], false, false);
        acc[mi][nt] = __builtin_amdgcn_wmma_f32_16x16x32_bf16(
            false, alo[mi].v, false, bhi.v, (short)0, acc[mi][nt], false, false);
        acc[mi][nt] = __builtin_amdgcn_wmma_f32_16x16x32_bf16(
            false, ahi[mi].v, false, blo.v, (short)0, acc[mi][nt], false, false);
      }
    }
  };

  // ---- software pipeline: global loads for kt+1 overlap compute on kt
  gload(0);
  lstore(0);
  __syncthreads();

  for (int kt = 0; kt < KTILES; ++kt) {
    const int cur = kt & 1;
    if (kt + 1 < KTILES) gload(kt + 1);
    if (kt + 2 < KTILES) {  // pull tile kt+2 toward L2/WGP$ (global_prefetch_b8)
      __builtin_prefetch(xg + (kt + 2) * BK, 0, 1);
      __builtin_prefetch(wg + (kt + 2) * BK, 0, 1);
    }
    compute(cur);
    if (kt + 1 < KTILES) lstore(cur ^ 1);
    __syncthreads();  // single barrier: double-buffer hazard separation
  }

  // ---- epilogue: bias add + fp32 store (C/D layout: VGPR r -> M=r / M=r+8)
#pragma unroll
  for (int nt = 0; nt < 4; ++nt) {
    const int n = n0 + wn * 64 + nt * 16 + (lane & 15);
    const float bias = Bias[n];
#pragma unroll
    for (int mi = 0; mi < 2; ++mi) {
      const int mbase = m0 + wm * 32 + mi * 16 + ((lane >> 4) << 3);
      float* o = Out + (size_t)mbase * DIM_N + n;
#pragma unroll
      for (int r = 0; r < 8; ++r)
        o[(size_t)r * DIM_N] = acc[mi][nt][r] + bias;
    }
  }
}

extern "C" void kernel_launch(void* const* d_in, const int* in_sizes, int n_in,
                              void* d_out, int out_size, void* d_ws, size_t ws_size,
                              hipStream_t stream) {
  (void)in_sizes; (void)n_in; (void)out_size; (void)d_ws; (void)ws_size;
  const float* x = (const float*)d_in[0];   // [4, 2048, 2048] f32
  const float* W = (const float*)d_in[1];   // [2048, 2048]   f32
  const float* b = (const float*)d_in[2];   // [2048]         f32
  float* out = (float*)d_out;               // [8192, 2048]   f32 (size-1 dim is free)

  dim3 grid(DIM_N / BN, M_TOTAL / BM);      // (16, 64)
  dim3 block(256);                          // 8 wave32 per block
  ScratchLinear_wmma_bf16x3<<<grid, block, 0, stream>>>(x, W, b, out);
}
